// MercuryMoE_56693568307386
// MI455X (gfx1250) — compile-verified, matrix operand order checked by reading, not run on hardware
//
#include <hip/hip_runtime.h>
#include <hip/hip_bf16.h>
#include <math.h>

// ---------------- model constants ----------------
#define DIMW   384
#define HEADS  6
#define HD     64
#define LAYERS 12
#define EXPERTS 8
#define HID    1536
#define NTOK   785
#define BATCH  8
#define TTOK   (BATCH*NTOK)      // 6280
#define NCLS   101
#define CONVM  (BATCH*784)       // 6272
#define CONVK  1536

typedef __attribute__((ext_vector_type(16))) _Float16 v16h;
typedef __attribute__((ext_vector_type(8)))  float    v8f;

#if defined(__has_builtin) && __has_builtin(__builtin_amdgcn_sched_barrier)
#define SCHED_FENCE() __builtin_amdgcn_sched_barrier(0)
#else
#define SCHED_FENCE() ((void)0)
#endif

__device__ __forceinline__ float gelu_exact(float x) {
    return 0.5f * x * (1.0f + erff(x * 0.70710678118654752f));
}

__device__ __forceinline__ v16h pack16(const float4& x0, const float4& x1,
                                       const float4& x2, const float4& x3) {
    v16h r;
    r[0]  = (_Float16)x0.x; r[1]  = (_Float16)x0.y;
    r[2]  = (_Float16)x0.z; r[3]  = (_Float16)x0.w;
    r[4]  = (_Float16)x1.x; r[5]  = (_Float16)x1.y;
    r[6]  = (_Float16)x1.z; r[7]  = (_Float16)x1.w;
    r[8]  = (_Float16)x2.x; r[9]  = (_Float16)x2.y;
    r[10] = (_Float16)x2.z; r[11] = (_Float16)x2.w;
    r[12] = (_Float16)x3.x; r[13] = (_Float16)x3.y;
    r[14] = (_Float16)x3.z; r[15] = (_Float16)x3.w;
    return r;
}

// ---------------- generic WMMA GEMM ----------------
// One wave32 computes a 16x64 C tile (4 x 16x16 WMMA subtiles, K-step 32).
// C = epilogue(A[MxK] @ B[KxN] + bias).
// A row-major (lda).  BT==0 -> B[k*ldb+n] (K-major); BT==1 -> B[n*ldb+k].
// MODE: 0 = store (acc+bias)
//       1 = store gelu(acc+bias)
//       2 = C += (acc+bias)                       (residual accumulate)
//       3 = C += scale[m*sstride] * (acc+bias)    (MoE weighted accumulate)
// K must be a multiple of 32 (384 / 1536 / 192 all are).
// M/N edges handled by CLAMPED loads (garbage only reaches C rows/cols whose
// stores are guarded) so the K-loop is branch-free.
// sched_barrier(0) between the load phase and the convert phase pins ALL
// slab loads above ALL consumers -> ~20 loads in flight per wave instead of
// the scheduler's default load/wait/convert lockstep.
template <int BT, int MODE>
__global__ __launch_bounds__(32)
void wmma_gemm_k(const float* __restrict__ A, int lda,
                 const float* __restrict__ B, int ldb,
                 const float* __restrict__ bias,
                 float* __restrict__ C, int ldc,
                 int M, int N, int K,
                 const float* __restrict__ scale, int sstride) {
    const int lane = threadIdx.x;
    const int half = lane >> 4;
    const int l16  = lane & 15;
    const int m0   = blockIdx.y * 16;
    const int n0   = blockIdx.x * 64;

    const int m_a = min(m0 + l16, M - 1);       // clamped A row for this lane
    int n_b[4];
#pragma unroll
    for (int s = 0; s < 4; ++s) n_b[s] = min(n0 + s * 16 + l16, N - 1);

    v8f acc[4];
#pragma unroll
    for (int s = 0; s < 4; ++s) acc[s] = (v8f){};

    const float* arow = A + (size_t)m_a * lda;

    for (int k0 = 0; k0 < K; k0 += 32) {
        if (k0 + 32 < K) {
            __builtin_prefetch(arow + k0 + 32, 0, 1);               // next A slab
            if (BT == 0)
                __builtin_prefetch(&B[(size_t)(k0 + 32 + lane) * ldb + n0], 0, 1);
            else
                __builtin_prefetch(&B[(size_t)n_b[lane & 3] * ldb + k0 + 32], 0, 1);
        }

        // ================= phase 1: issue ALL loads =================
        // A operand rows: k = 8*half + {0..7} and 16 + 8*half + {0..7}
        const float* ap = arow + k0 + 8 * half;
        float4 araw[4];
        araw[0] = *(const float4*)(ap);
        araw[1] = *(const float4*)(ap + 4);
        araw[2] = *(const float4*)(ap + 16);
        araw[3] = *(const float4*)(ap + 20);

        float4 braw4[4][4];        // BT==1 path
        float  braws[4][16];       // BT==0 path
        if (BT) {
#pragma unroll
            for (int s = 0; s < 4; ++s) {
                const float* bp = B + (size_t)n_b[s] * ldb + k0 + 8 * half;
                braw4[s][0] = *(const float4*)(bp);
                braw4[s][1] = *(const float4*)(bp + 4);
                braw4[s][2] = *(const float4*)(bp + 16);
                braw4[s][3] = *(const float4*)(bp + 20);
            }
        } else {
            // lanes 0..15 of each half read 16 consecutive columns of one
            // B row -> 64B coalesced per element index, no divergence
#pragma unroll
            for (int s = 0; s < 4; ++s) {
                const float* bp = B + (size_t)(k0 + 8 * half) * ldb + n_b[s];
#pragma unroll
                for (int i = 0; i < 8; ++i) {
                    braws[s][i]     = bp[(size_t)i * ldb];
                    braws[s][8 + i] = bp[(size_t)(16 + i) * ldb];
                }
            }
        }

        // pin: nothing below may move above (keeps all loads batched & live)
        SCHED_FENCE();

        // ================= phase 2: convert =================
        const v16h av = pack16(araw[0], araw[1], araw[2], araw[3]);
        v16h bv[4];
#pragma unroll
        for (int s = 0; s < 4; ++s) {
            if (BT) {
                bv[s] = pack16(braw4[s][0], braw4[s][1], braw4[s][2], braw4[s][3]);
            } else {
#pragma unroll
                for (int i = 0; i < 16; ++i) bv[s][i] = (_Float16)braws[s][i];
            }
        }

        SCHED_FENCE();

        // ================= phase 3: matrix ops =================
#pragma unroll
        for (int s = 0; s < 4; ++s)
            acc[s] = __builtin_amdgcn_wmma_f32_16x16x32_f16(
                false, av, false, bv[s], (short)0, acc[s], false, false);
    }

    // ---- epilogue:  C rows = m0 + r + 8*half, cols = n0 + s*16 + l16 ----
#pragma unroll
    for (int s = 0; s < 4; ++s) {
#pragma unroll
        for (int r = 0; r < 8; ++r) {
            const int m = m0 + r + 8 * half;
            const int n = n0 + s * 16 + l16;
            if (m < M && n < N) {
                float val = acc[s][r] + (bias ? bias[n] : 0.0f);
                float* cp = &C[(size_t)m * ldc + n];
                if (MODE == 0)      *cp = val;
                else if (MODE == 1) *cp = gelu_exact(val);
                else if (MODE == 2) *cp = *cp + val;
                else                *cp = *cp + scale[(size_t)m * sstride] * val;
            }
        }
    }
}

// ---------------- im2col for the tube conv ----------------
// A_col[m][k], m = b*784 + ti*196 + py*14 + px,  k = c*512 + dt*256 + ky*16 + kx
__global__ void im2col_k(const float* __restrict__ x, float* __restrict__ acol) {
    long idx = (long)blockIdx.x * 256 + threadIdx.x;
    const long total = (long)CONVM * CONVK;
    if (idx >= total) return;
    int kk = (int)(idx % CONVK);
    int m  = (int)(idx / CONVK);
    int b = m / 784, jj = m % 784;
    int ti = jj / 196, r = jj % 196, py = r / 14, px = r % 14;
    int c = kk / 512, rem = kk % 512, dt = rem / 256, rr = rem % 256;
    int ky = rr / 16, kx = rr % 16;
    long src = ((((long)(b * 3 + c) * 8 + (ti * 2 + dt)) * 224 + (py * 16 + ky)) * 224
               + (px * 16 + kx));
    acol[idx] = x[src];
}

// ---------------- embed: cls + pos over conv tokens ----------------
__global__ void embed_k(const float* __restrict__ conv, const float* __restrict__ cls,
                        const float* __restrict__ pos, float* __restrict__ t) {
    int idx = blockIdx.x * 256 + threadIdx.x;
    const int total = TTOK * DIMW;
    if (idx >= total) return;
    int d = idx % DIMW;
    int tok = (idx / DIMW) % NTOK;
    int b = idx / (DIMW * NTOK);
    float v = (tok == 0) ? cls[d]
                         : conv[((size_t)(b * 784 + (tok - 1))) * DIMW + d];
    t[idx] = v + pos[(size_t)tok * DIMW + d];
}

// ---------------- LayerNorm: one wave per token ----------------
__global__ __launch_bounds__(32)
void ln_k(const float* __restrict__ x, const float* __restrict__ w,
          const float* __restrict__ b, float* __restrict__ y) {
    const int tok = blockIdx.x;
    const int lane = threadIdx.x;
    const float* xp = x + (size_t)tok * DIMW;
    float* yp = y + (size_t)tok * DIMW;
    float vals[12];
    float s = 0.f, s2 = 0.f;
#pragma unroll
    for (int i = 0; i < 12; ++i) {
        float v = xp[lane + i * 32];
        vals[i] = v; s += v; s2 += v * v;
    }
#pragma unroll
    for (int sh = 16; sh > 0; sh >>= 1) {
        s  += __shfl_xor(s,  sh, 32);
        s2 += __shfl_xor(s2, sh, 32);
    }
    const float mu  = s * (1.0f / DIMW);
    const float var = s2 * (1.0f / DIMW) - mu * mu;
    const float rs  = rsqrtf(var + 1e-5f);
#pragma unroll
    for (int i = 0; i < 12; ++i) {
        int d = lane + i * 32;
        yp[d] = (vals[i] - mu) * rs * w[d] + b[d];
    }
}

// ---------------- attention: one wave per (b, h, query), online softmax ----------------
__global__ __launch_bounds__(32)
void attn_k(const float* __restrict__ q, const float* __restrict__ k,
            const float* __restrict__ v, float* __restrict__ o) {
    const int gid = blockIdx.x;              // b*HEADS*NTOK + h*NTOK + qi
    const int qi = gid % NTOK;
    const int h  = (gid / NTOK) % HEADS;
    const int b  = gid / (NTOK * HEADS);
    const int lane = threadIdx.x;
    const size_t rowq = (size_t)(b * NTOK + qi);

    float qv[HD];
    const float* qp = q + rowq * DIMW + h * HD;
#pragma unroll
    for (int d = 0; d < HD; ++d) qv[d] = qp[d];

    float m_i = -1e30f, l_i = 0.0f;
    float ov[HD];
#pragma unroll
    for (int d = 0; d < HD; ++d) ov[d] = 0.0f;

    for (int kk = lane; kk < NTOK; kk += 32) {
        const size_t rowk = (size_t)(b * NTOK + kk);
        const float* kp = k + rowk * DIMW + h * HD;
        float s = 0.0f;
#pragma unroll
        for (int d = 0; d < HD; ++d) s += qv[d] * kp[d];
        s *= 0.125f;                                  // 1/sqrt(64)
        const float mnew = fmaxf(m_i, s);
        const float corr = __expf(m_i - mnew);
        const float p    = __expf(s - mnew);
        l_i = l_i * corr + p;
        const float* vp = v + rowk * DIMW + h * HD;
#pragma unroll
        for (int d = 0; d < HD; ++d) ov[d] = ov[d] * corr + p * vp[d];
        m_i = mnew;
    }
    // cross-lane combine
    float mmax = m_i;
#pragma unroll
    for (int sh = 16; sh > 0; sh >>= 1) mmax = fmaxf(mmax, __shfl_xor(mmax, sh, 32));
    const float c = __expf(m_i - mmax);
    float lsum = l_i * c;
#pragma unroll
    for (int sh = 16; sh > 0; sh >>= 1) lsum += __shfl_xor(lsum, sh, 32);
    const float inv = 1.0f / lsum;

    float* op = o + rowq * DIMW + h * HD;
#pragma unroll
    for (int d = 0; d < HD; ++d) {
        float val = ov[d] * c;
#pragma unroll
        for (int sh = 16; sh > 0; sh >>= 1) val += __shfl_xor(val, sh, 32);
        if (lane == 0) op[d] = val * inv;
    }
}

// ---------------- MoE gate: softmax + top2 (lowest-index ties like jax top_k) -----------
__global__ void gate_k(const float* __restrict__ glog, float* __restrict__ probs,
                       float* __restrict__ wsel, float* __restrict__ selb, int T) {
    int t = blockIdx.x * 256 + threadIdx.x;
    if (t >= T) return;
    float l[EXPERTS];
    float mx = -1e30f;
#pragma unroll
    for (int e = 0; e < EXPERTS; ++e) { l[e] = glog[(size_t)t * EXPERTS + e]; mx = fmaxf(mx, l[e]); }
    float s = 0.f;
#pragma unroll
    for (int e = 0; e < EXPERTS; ++e) { l[e] = __expf(l[e] - mx); s += l[e]; }
    const float inv = 1.0f / s;
    float p[EXPERTS];
#pragma unroll
    for (int e = 0; e < EXPERTS; ++e) { p[e] = l[e] * inv; probs[(size_t)t * EXPERTS + e] = p[e]; }
    int i1 = 0;
#pragma unroll
    for (int e = 1; e < EXPERTS; ++e) if (p[e] > p[i1]) i1 = e;
    int i2 = (i1 == 0) ? 1 : 0;
#pragma unroll
    for (int e = 0; e < EXPERTS; ++e) if (e != i1 && e != i2 && p[e] > p[i2]) i2 = e;
    const float inv2 = 1.0f / (p[i1] + p[i2]);
#pragma unroll
    for (int e = 0; e < EXPERTS; ++e) {
        wsel[(size_t)t * EXPERTS + e] = 0.0f;
        selb[(size_t)t * EXPERTS + e] = 0.0f;
    }
    wsel[(size_t)t * EXPERTS + i1] = p[i1] * inv2;
    wsel[(size_t)t * EXPERTS + i2] = p[i2] * inv2;
    selb[(size_t)t * EXPERTS + i1] = 1.0f;
    selb[(size_t)t * EXPERTS + i2] = 1.0f;
}

// ---------------- aux loss: deterministic single-block tree reduction ----------------
__global__ __launch_bounds__(256)
void aux_k(const float* __restrict__ probs, const float* __restrict__ selb,
           float* __restrict__ aux, int T) {
    __shared__ float sf[256 * EXPERTS];
    __shared__ float sp[256 * EXPERTS];
    const int tid = threadIdx.x;
    float fl[EXPERTS], pl[EXPERTS];
#pragma unroll
    for (int e = 0; e < EXPERTS; ++e) { fl[e] = 0.f; pl[e] = 0.f; }
    for (int t = tid; t < T; t += 256)
#pragma unroll
        for (int e = 0; e < EXPERTS; ++e) {
            fl[e] += selb[(size_t)t * EXPERTS + e];
            pl[e] += probs[(size_t)t * EXPERTS + e];
        }
#pragma unroll
    for (int e = 0; e < EXPERTS; ++e) { sf[tid * EXPERTS + e] = fl[e]; sp[tid * EXPERTS + e] = pl[e]; }
    __syncthreads();
    for (int s = 128; s > 0; s >>= 1) {
        if (tid < s)
#pragma unroll
            for (int e = 0; e < EXPERTS; ++e) {
                sf[tid * EXPERTS + e] += sf[(tid + s) * EXPERTS + e];
                sp[tid * EXPERTS + e] += sp[(tid + s) * EXPERTS + e];
            }
        __syncthreads();
    }
    if (tid == 0) {
        float a = 0.f;
        const float invT = 1.0f / (float)T;
#pragma unroll
        for (int e = 0; e < EXPERTS; ++e) a += (sf[e] * invT) * (sp[e] * invT);
        aux[0] += (float)EXPERTS * a;
    }
}

__global__ void zero_aux_k(float* aux) { if (threadIdx.x == 0 && blockIdx.x == 0) aux[0] = 0.0f; }

__global__ void cls_gather_k(const float* __restrict__ xn, float* __restrict__ clsb) {
    int i = blockIdx.x * 256 + threadIdx.x;
    if (i >= BATCH * DIMW) return;
    int b = i / DIMW, d = i % DIMW;
    clsb[i] = xn[(size_t)b * NTOK * DIMW + d];
}

__global__ void finalize_k(const float* __restrict__ aux, float* __restrict__ out) {
    if (threadIdx.x == 0 && blockIdx.x == 0) out[BATCH * NCLS] = aux[0] * 0.01f;
}

// ======================= host side =======================
static inline void gemm(hipStream_t s, const float* A, int lda,
                        const float* B, int ldb, int btrans, const float* bias,
                        float* C, int ldc, int M, int N, int K,
                        int mode, const float* scale = nullptr, int sstride = 0) {
    dim3 g((N + 63) / 64, (M + 15) / 16), blk(32);
    if (btrans) {
        wmma_gemm_k<1, 0><<<g, blk, 0, s>>>(A, lda, B, ldb, bias, C, ldc, M, N, K, scale, sstride);
        return;
    }
    switch (mode) {
    case 0: wmma_gemm_k<0, 0><<<g, blk, 0, s>>>(A, lda, B, ldb, bias, C, ldc, M, N, K, scale, sstride); break;
    case 1: wmma_gemm_k<0, 1><<<g, blk, 0, s>>>(A, lda, B, ldb, bias, C, ldc, M, N, K, scale, sstride); break;
    case 2: wmma_gemm_k<0, 2><<<g, blk, 0, s>>>(A, lda, B, ldb, bias, C, ldc, M, N, K, scale, sstride); break;
    default: wmma_gemm_k<0, 3><<<g, blk, 0, s>>>(A, lda, B, ldb, bias, C, ldc, M, N, K, scale, sstride); break;
    }
}

extern "C" void kernel_launch(void* const* d_in, const int* in_sizes, int n_in,
                              void* d_out, int out_size, void* d_ws, size_t ws_size,
                              hipStream_t stream) {
    (void)in_sizes; (void)n_in; (void)out_size; (void)ws_size;
    // Input order: x, then params pytree-flattened (sorted dict keys):
    // blocks[0..11] (MoE layers 18 leaves, dense layers 16), then
    // cls, h_b1, h_b2, h_w1, h_w2, ln_b, ln_w, pos, tok_b, tok_w.
    auto F = [&](int i) { return (const float*)d_in[i]; };
    const float* x = F(0);
    const int top = 1 + 6 * 18 + 6 * 16;          // 205
    const float* cls_p  = F(top + 0);
    const float* h_b1   = F(top + 1);
    const float* h_b2   = F(top + 2);
    const float* h_w1   = F(top + 3);
    const float* h_w2   = F(top + 4);
    const float* ln_b   = F(top + 5);
    const float* ln_w   = F(top + 6);
    const float* pos    = F(top + 7);
    const float* tok_b  = F(top + 8);
    const float* tok_w  = F(top + 9);

    // workspace layout (f32)
    float* W = (float*)d_ws;
    const size_t TD = (size_t)TTOK * DIMW;        // 2,411,520
    float* t    = W;
    float* xn   = t  + TD;
    float* qb   = xn + TD;                        // also conv output buffer
    float* kb   = qb + TD;
    float* vb   = kb + TD;
    float* ao   = vb + TD;
    float* hb   = ao + TD;                        // T*HID (also im2col buffer)
    float* glog = hb + (size_t)TTOK * HID;
    float* probs= glog + (size_t)TTOK * EXPERTS;
    float* wsel = probs+ (size_t)TTOK * EXPERTS;
    float* selb = wsel + (size_t)TTOK * EXPERTS;
    float* auxb = selb + (size_t)TTOK * EXPERTS;
    float* clsb = auxb + 16;
    float* h1b  = clsb + BATCH * DIMW;

    zero_aux_k<<<1, 32, 0, stream>>>(auxb);

    // --- tokenizer: im2col + WMMA GEMM (tok_w is [DIM][1536] row-major -> BT=1) ---
    {
        long total = (long)CONVM * CONVK;
        im2col_k<<<(unsigned)((total + 255) / 256), 256, 0, stream>>>(x, hb);
        gemm(stream, hb, CONVK, tok_w, CONVK, /*btrans=*/1, tok_b,
             qb, DIMW, CONVM, DIMW, CONVK, /*mode=*/0);
        embed_k<<<(TTOK * DIMW + 255) / 256, 256, 0, stream>>>(qb, cls_p, pos, t);
    }

    for (int i = 0; i < LAYERS; ++i) {
        const int off = 1 + (i / 2) * 34 + (i % 2) * 18;
        const bool moe = (i % 2 == 0);
        const float *bk, *bo, *bq, *bv_, *ln1b, *ln1w, *ln2b, *ln2w, *wk, *wo, *wq, *wv;
        bk  = F(off + 0); bo = F(off + 1); bq = F(off + 2); bv_ = F(off + 3);
        if (moe) {
            ln1b = F(off + 10); ln1w = F(off + 11); ln2b = F(off + 12); ln2w = F(off + 13);
            wk = F(off + 14); wo = F(off + 15); wq = F(off + 16); wv = F(off + 17);
        } else {
            ln1b = F(off + 8);  ln1w = F(off + 9);  ln2b = F(off + 10); ln2w = F(off + 11);
            wk = F(off + 12); wo = F(off + 13); wq = F(off + 14); wv = F(off + 15);
        }

        // attention
        ln_k<<<TTOK, 32, 0, stream>>>(t, ln1w, ln1b, xn);
        gemm(stream, xn, DIMW, wq, DIMW, 0, bq, qb, DIMW, TTOK, DIMW, DIMW, 0);
        gemm(stream, xn, DIMW, wk, DIMW, 0, bk, kb, DIMW, TTOK, DIMW, DIMW, 0);
        gemm(stream, xn, DIMW, wv, DIMW, 0, bv_, vb, DIMW, TTOK, DIMW, DIMW, 0);
        attn_k<<<BATCH * HEADS * NTOK, 32, 0, stream>>>(qb, kb, vb, ao);
        gemm(stream, ao, DIMW, wo, DIMW, 0, bo, t, DIMW, TTOK, DIMW, DIMW, /*accum*/2);

        // FFN
        ln_k<<<TTOK, 32, 0, stream>>>(t, ln2w, ln2b, xn);
        if (moe) {
            const float* e_b1 = F(off + 4);
            const float* e_b2 = F(off + 5);
            const float* e_w1 = F(off + 6);
            const float* e_w2 = F(off + 7);
            const float* gate_b = F(off + 8);
            const float* gate_w = F(off + 9);
            gemm(stream, xn, DIMW, gate_w, EXPERTS, 0, gate_b, glog, EXPERTS,
                 TTOK, EXPERTS, DIMW, 0);
            gate_k<<<(TTOK + 255) / 256, 256, 0, stream>>>(glog, probs, wsel, selb, TTOK);
            aux_k<<<1, 256, 0, stream>>>(probs, selb, auxb, TTOK);
            for (int e = 0; e < EXPERTS; ++e) {
                const float* w1 = e_w1 + (size_t)e * DIMW * HID;
                const float* b1 = e_b1 + (size_t)e * HID;
                const float* w2 = e_w2 + (size_t)e * HID * DIMW;
                const float* b2 = e_b2 + (size_t)e * DIMW;
                gemm(stream, xn, DIMW, w1, HID, 0, b1, hb, HID, TTOK, HID, DIMW, /*gelu*/1);
                gemm(stream, hb, HID, w2, DIMW, 0, b2, t, DIMW, TTOK, DIMW, HID,
                     /*scaled accum*/3, wsel + e, EXPERTS);
            }
        } else {
            const float* f_b1 = F(off + 4);
            const float* f_b2 = F(off + 5);
            const float* f_w1 = F(off + 6);
            const float* f_w2 = F(off + 7);
            gemm(stream, xn, DIMW, f_w1, HID, 0, f_b1, hb, HID, TTOK, HID, DIMW, /*gelu*/1);
            gemm(stream, hb, HID, f_w2, DIMW, 0, f_b2, t, DIMW, TTOK, DIMW, HID, /*accum*/2);
        }
    }

    // head
    ln_k<<<TTOK, 32, 0, stream>>>(t, ln_w, ln_b, xn);
    cls_gather_k<<<(BATCH * DIMW + 255) / 256, 256, 0, stream>>>(xn, clsb);
    gemm(stream, clsb, DIMW, h_w1, DIMW / 2, 0, h_b1, h1b, DIMW / 2,
         BATCH, DIMW / 2, DIMW, /*gelu*/1);
    gemm(stream, h1b, DIMW / 2, h_w2, NCLS, 0, h_b2, (float*)d_out, NCLS,
         BATCH, NCLS, DIMW / 2, /*store*/0);
    finalize_k<<<1, 32, 0, stream>>>(auxb, (float*)d_out);
}
